// TableAdapterLayer_55327768707128
// MI455X (gfx1250) — matrix operand order repaired and checked by reading, not compiled, not run on hardware
//
#include <hip/hip_runtime.h>
#include <hip/hip_bf16.h>
#include <math.h>

#define B_   8
#define S_   512
#define H_   1024
#define FF_  3072
#define L_   256
#define NH_  16
#define DH_  64
#define NTOK (B_ * S_)   // 4096

typedef __attribute__((ext_vector_type(16))) _Float16 v16h;
typedef __attribute__((ext_vector_type(8)))  _Float16 v8h;
typedef __attribute__((ext_vector_type(8)))  float    v8f;

// ---------------------------------------------------------------------------
// Helpers
// ---------------------------------------------------------------------------
__device__ __forceinline__ float gelu_exact(float x) {
  return 0.5f * x * (1.0f + erff(x * 0.70710678118654752f));
}

__device__ __forceinline__ v8f wmma16(v16h a, v16h b, v8f c) {
  // v_wmma_f32_16x16x32_f16  D = A(16x32 f16) * B(32x16 f16) + C(16x16 f32)
  return __builtin_amdgcn_wmma_f32_16x16x32_f16(
      /*neg_a=*/false, a, /*neg_b=*/false, b,
      /*c_mod=*/(short)0, c, /*reuse_a=*/false, /*reuse_b=*/false);
}

// A-matrix 16x32 f16 fragment (ISA 7.12.2): lane m=lane&15, h=lane>>4,
// halves[0..7] = K (h*8 .. h*8+7), halves[8..15] = K (16+h*8 .. 16+h*8+7)
template <typename T>
__device__ __forceinline__ v16h load_a_frag(const T* base, int lda, int lane) {
  const int m = lane & 15, h = lane >> 4;
  const T* p = base + (size_t)m * lda + h * 8;
  v8h lo = *(const v8h*)(p);
  v8h hi = *(const v8h*)(p + 16);
  union { v16h v; v8h h2[2]; } u; u.h2[0] = lo; u.h2[1] = hi;
  return u.v;
}

// B-matrix 32x16 f16 fragment from Bt stored [N][K]: lane n=lane&15, h=lane>>4,
// halves[0..7] = K (h*8 .. h*8+7), halves[8..15] = K (16+h*8 .. 16+h*8+7) for column n.
template <typename T>
__device__ __forceinline__ v16h load_b_frag(const T* base, int ldb, int lane) {
  const int n = lane & 15, h = lane >> 4;
  const T* p = base + (size_t)n * ldb + h * 8;
  v8h lo = *(const v8h*)(p);
  v8h hi = *(const v8h*)(p + 16);
  union { v16h v; v8h h2[2]; } u; u.h2[0] = lo; u.h2[1] = hi;
  return u.v;
}

// ---------------------------------------------------------------------------
// Generic WMMA GEMM: out[M,N] = epilogue(A[M,K](f16) * Wt[N,K](f16)^T + bias)
// flags: bit0 = GELU, bit1 = f16 output (else f32)
// Block: 128 threads (4 waves), tile 64x64; wave tile 32x32 (2x2 WMMA tiles)
// ---------------------------------------------------------------------------
__global__ void __launch_bounds__(128)
gemm_wmma_kernel(const _Float16* __restrict__ A, const _Float16* __restrict__ Wt,
                 const float* __restrict__ bias, void* __restrict__ outp,
                 int M, int N, int K, int flags) {
  const int lane = threadIdx.x & 31;
  const int wave = threadIdx.x >> 5;
  const int row0 = blockIdx.x * 64 + (wave >> 1) * 32;
  const int col0 = blockIdx.y * 64 + (wave & 1) * 32;

  const _Float16* arow0 = A  + (size_t)row0 * K;
  const _Float16* arow1 = A  + (size_t)(row0 + 16) * K;
  const _Float16* bcol0 = Wt + (size_t)col0 * K;
  const _Float16* bcol1 = Wt + (size_t)(col0 + 16) * K;

  v8f acc00 = {}, acc01 = {}, acc10 = {}, acc11 = {};
  for (int kk = 0; kk < K; kk += 32) {
    __builtin_prefetch(arow0 + kk + 256, 0, 0);  // global_prefetch_b8
    __builtin_prefetch(bcol0 + kk + 256, 0, 0);
    v16h a0 = load_a_frag(arow0 + kk, K, lane);
    v16h a1 = load_a_frag(arow1 + kk, K, lane);
    v16h b0 = load_b_frag(bcol0 + kk, K, lane);
    v16h b1 = load_b_frag(bcol1 + kk, K, lane);
    acc00 = wmma16(a0, b0, acc00);
    acc01 = wmma16(a0, b1, acc01);
    acc10 = wmma16(a1, b0, acc10);
    acc11 = wmma16(a1, b1, acc11);
  }

  const int n  = lane & 15;
  const int hi = lane >> 4;
  auto store_tile = [&](v8f acc, int rbase, int cbase) {
    for (int r = 0; r < 8; ++r) {
      const int m = rbase + r + 8 * hi;   // C layout: VGPR r -> M = r + 8*hi
      const int c = cbase + n;
      float v = acc[r];
      if (bias) v += bias[c];
      if (flags & 1) v = gelu_exact(v);
      if (flags & 2) ((_Float16*)outp)[(size_t)m * N + c] = (_Float16)v;
      else           ((float*)outp)[(size_t)m * N + c] = v;
    }
  };
  store_tile(acc00, row0,      col0);
  store_tile(acc01, row0,      col0 + 16);
  store_tile(acc10, row0 + 16, col0);
  store_tile(acc11, row0 + 16, col0 + 16);
}

// ---------------------------------------------------------------------------
// Attention: one wave per (b, head, 16-row q tile). Scores via WMMA into LDS,
// softmax in LDS, probs (f16, LDS) x Vt via WMMA -> ctx.
// ---------------------------------------------------------------------------
__global__ void __launch_bounds__(32)
attention_kernel(const _Float16* __restrict__ Q, const _Float16* __restrict__ Km,
                 const _Float16* __restrict__ Vt, const float* __restrict__ smask,
                 _Float16* __restrict__ ctx) {
  __shared__ float    sc[16 * S_];        // 32 KB
  __shared__ _Float16 pr[16 * S_];        // 16 KB

  const int blk = blockIdx.x;
  const int qt  = blk % (S_ / 16);
  const int hh  = (blk / (S_ / 16)) % NH_;
  const int bb  = blk / ((S_ / 16) * NH_);
  const int lane = threadIdx.x;
  const int n  = lane & 15;
  const int hi = lane >> 4;
  const int q0 = qt * 16;

  // Q fragments for this q tile (two 32-wide chunks over DH=64)
  const _Float16* qbase = Q + (size_t)(bb * S_ + q0) * H_ + hh * DH_;
  v16h qa0 = load_a_frag(qbase,      H_, lane);
  v16h qa1 = load_a_frag(qbase + 32, H_, lane);

  // scores = Q K^T / sqrt(DH) + (1-mask)*-1e4
  for (int k0 = 0; k0 < S_; k0 += 16) {
    const _Float16* kbase = Km + (size_t)(bb * S_ + k0) * H_ + hh * DH_;
    v16h kb0 = load_b_frag(kbase,      H_, lane);
    v16h kb1 = load_b_frag(kbase + 32, H_, lane);
    v8f acc = {};
    acc = wmma16(qa0, kb0, acc);
    acc = wmma16(qa1, kb1, acc);
    for (int r = 0; r < 8; ++r) {
      const int m = r + 8 * hi;
      const int kk = k0 + n;
      const float msk = smask[((size_t)bb * S_ + q0 + m) * S_ + kk];
      sc[m * S_ + kk] = acc[r] * 0.125f + (1.0f - msk) * (-10000.0f);
    }
  }
  __syncthreads();

  // softmax per row (16 rows, lanes 0..15)
  if (lane < 16) {
    float mx = -3.0e38f;
    for (int k = 0; k < S_; ++k) mx = fmaxf(mx, sc[lane * S_ + k]);
    float sum = 0.0f;
    for (int k = 0; k < S_; ++k) {
      float e = __expf(sc[lane * S_ + k] - mx);
      sc[lane * S_ + k] = e;
      sum += e;
    }
    const float inv = 1.0f / sum;
    for (int k = 0; k < S_; ++k)
      pr[lane * S_ + k] = (_Float16)(sc[lane * S_ + k] * inv);
  }
  __syncthreads();

  // ctx = probs @ V   (A fragments read from LDS -> ds_load; B from Vt[d][s])
  for (int d0 = 0; d0 < DH_; d0 += 16) {
    const _Float16* vtbase = Vt + ((size_t)(bb * NH_ + hh) * DH_ + d0) * S_;
    v8f acc = {};
    for (int kk = 0; kk < S_; kk += 32) {
      v16h a = load_a_frag((const _Float16*)pr + kk, S_, lane);
      v16h b = load_b_frag(vtbase + kk, S_, lane);
      acc = wmma16(a, b, acc);
    }
    for (int r = 0; r < 8; ++r) {
      const int m = r + 8 * hi;
      ctx[(size_t)(bb * S_ + q0 + m) * H_ + hh * DH_ + d0 + n] = (_Float16)acc[r];
    }
  }
}

// ---------------------------------------------------------------------------
// Elementwise / reduction kernels
// ---------------------------------------------------------------------------
__global__ void transpose_cast_kernel(const float* __restrict__ in,  // [K,N]
                                      _Float16* __restrict__ out,    // [N,K]
                                      int K, int N) {
  size_t idx = (size_t)blockIdx.x * blockDim.x + threadIdx.x;
  size_t total = (size_t)K * N;
  if (idx >= total) return;
  int k = (int)(idx / N), n = (int)(idx % N);
  out[(size_t)n * K + k] = (_Float16)in[idx];
}

__global__ void concat2_kernel(const float* __restrict__ a, const float* __restrict__ b,
                               _Float16* __restrict__ cat2, _Float16* __restrict__ cat4) {
  size_t idx = (size_t)blockIdx.x * blockDim.x + threadIdx.x;
  if (idx >= (size_t)NTOK * H_) return;
  size_t row = idx / H_, j = idx % H_;
  _Float16 av = (_Float16)a[idx], bv = (_Float16)b[idx];
  cat2[row * (2 * H_) + j]       = av;
  cat2[row * (2 * H_) + H_ + j]  = bv;
  cat4[row * (4 * H_) + j]       = av;
  cat4[row * (4 * H_) + H_ + j]  = bv;
}

__global__ void zero_f32_kernel(float* __restrict__ p, size_t n) {
  size_t idx = (size_t)blockIdx.x * blockDim.x + threadIdx.x;
  if (idx < n) p[idx] = 0.0f;
}

__global__ void scatter_add_kernel(const float* __restrict__ seq, const int* __restrict__ ids,
                                   float* __restrict__ cell) {
  size_t idx = (size_t)blockIdx.x * blockDim.x + threadIdx.x;
  if (idx >= (size_t)NTOK * H_) return;
  int j = (int)(idx % H_);
  int tok = (int)(idx / H_);
  int b = tok / S_;
  int id = ids[tok];
  atomicAdd(&cell[((size_t)b * L_ + id) * H_ + j], seq[idx]);
}

__global__ void cast_f16_kernel(const float* __restrict__ in, _Float16* __restrict__ out, size_t n) {
  size_t idx = (size_t)blockIdx.x * blockDim.x + threadIdx.x;
  if (idx < n) out[idx] = (_Float16)in[idx];
}

__global__ void gather_kernel(const float* __restrict__ cellout, const int* __restrict__ ids,
                              const int* __restrict__ mask_ids, _Float16* __restrict__ cat4,
                              int col_off) {
  size_t idx = (size_t)blockIdx.x * blockDim.x + threadIdx.x;
  if (idx >= (size_t)NTOK * H_) return;
  int j = (int)(idx % H_);
  int tok = (int)(idx / H_);
  int b = tok / S_;
  int id = ids[tok];
  float m = (mask_ids[tok] == 0) ? 0.0f : 1.0f;
  cat4[(size_t)tok * (4 * H_) + col_off + j] =
      (_Float16)(cellout[((size_t)b * L_ + id) * H_ + j] * m);
}

__global__ void __launch_bounds__(256)
add_ln_kernel(const float* __restrict__ resid, const float* __restrict__ x,
              const float* __restrict__ g, const float* __restrict__ bta,
              float* __restrict__ out32, _Float16* __restrict__ out16) {
  __shared__ float s1[256], s2[256];
  const int row = blockIdx.x, tid = threadIdx.x;
  const float* r  = resid + (size_t)row * H_;
  const float* xx = x     + (size_t)row * H_;
  float a = 0.0f, b2 = 0.0f;
  for (int j = tid; j < H_; j += 256) {
    float v = r[j] + xx[j];
    a += v; b2 += v * v;
  }
  s1[tid] = a; s2[tid] = b2;
  __syncthreads();
  for (int st = 128; st > 0; st >>= 1) {
    if (tid < st) { s1[tid] += s1[tid + st]; s2[tid] += s2[tid + st]; }
    __syncthreads();
  }
  const float mean = s1[0] * (1.0f / H_);
  const float var  = s2[0] * (1.0f / H_) - mean * mean;
  const float inv  = rsqrtf(var + 1e-12f);
  for (int j = tid; j < H_; j += 256) {
    float v = (r[j] + xx[j] - mean) * inv * g[j] + bta[j];
    out32[(size_t)row * H_ + j] = v;
    if (out16) out16[(size_t)row * H_ + j] = (_Float16)v;
  }
}

__global__ void vtranspose_kernel(const _Float16* __restrict__ V, _Float16* __restrict__ Vt) {
  size_t idx = (size_t)blockIdx.x * blockDim.x + threadIdx.x;
  if (idx >= (size_t)B_ * NH_ * DH_ * S_) return;
  int s = (int)(idx % S_);
  int d = (int)((idx / S_) % DH_);
  int h = (int)((idx / ((size_t)S_ * DH_)) % NH_);
  int b = (int)(idx / ((size_t)S_ * DH_ * NH_));
  Vt[idx] = V[(size_t)(b * S_ + s) * H_ + h * DH_ + d];
}

// ---------------------------------------------------------------------------
// Host launcher
// ---------------------------------------------------------------------------
static inline dim3 ew_grid(size_t n) { return dim3((unsigned)((n + 255) / 256)); }

extern "C" void kernel_launch(void* const* d_in, const int* in_sizes, int n_in,
                              void* d_out, int out_size, void* d_ws, size_t ws_size,
                              hipStream_t stream) {
  const float* former = (const float*)d_in[0];
  const float* hidden = (const float*)d_in[1];
  const float* smask  = (const float*)d_in[2];
  const float* r_dpw  = (const float*)d_in[4];
  const float* r_dpb  = (const float*)d_in[5];
  const float* r_lw   = (const float*)d_in[6];
  const float* r_lb   = (const float*)d_in[7];
  const float* c_dpw  = (const float*)d_in[8];
  const float* c_dpb  = (const float*)d_in[9];
  const float* c_lw   = (const float*)d_in[10];
  const float* c_lb   = (const float*)d_in[11];
  const float* dp_w   = (const float*)d_in[12];
  const float* dp_b   = (const float*)d_in[13];
  const float* ln_g   = (const float*)d_in[14];
  const float* ln_b   = (const float*)d_in[15];
  const float* wq = (const float*)d_in[16]; const float* bq = (const float*)d_in[17];
  const float* wk = (const float*)d_in[18]; const float* bk = (const float*)d_in[19];
  const float* wv = (const float*)d_in[20]; const float* bv = (const float*)d_in[21];
  const float* wo = (const float*)d_in[22]; const float* bo = (const float*)d_in[23];
  const float* ln1_g = (const float*)d_in[24]; const float* ln1_b = (const float*)d_in[25];
  const float* w1 = (const float*)d_in[26]; const float* b1 = (const float*)d_in[27];
  const float* w2 = (const float*)d_in[28]; const float* b2 = (const float*)d_in[29];
  const float* ln2_g = (const float*)d_in[30]; const float* ln2_b = (const float*)d_in[31];
  const int* row_ids = (const int*)d_in[32];
  const int* col_ids = (const int*)d_in[33];

  char* ws = (char*)d_ws;
  size_t cur = 0;
  auto alloc = [&](size_t bytes) -> void* {
    cur = (cur + 255) & ~(size_t)255;
    void* p = ws + cur;
    cur += bytes;
    return p;
  };

  // f16 transposed weights [N][K]
  _Float16* wt_rdp = (_Float16*)alloc((size_t)H_ * 2 * H_ * 2);
  _Float16* wt_rl  = (_Float16*)alloc((size_t)H_ * H_ * 2);
  _Float16* wt_cdp = (_Float16*)alloc((size_t)H_ * 2 * H_ * 2);
  _Float16* wt_cl  = (_Float16*)alloc((size_t)H_ * H_ * 2);
  _Float16* wt_dp  = (_Float16*)alloc((size_t)H_ * 4 * H_ * 2);
  _Float16* wt_q   = (_Float16*)alloc((size_t)H_ * H_ * 2);
  _Float16* wt_k   = (_Float16*)alloc((size_t)H_ * H_ * 2);
  _Float16* wt_v   = (_Float16*)alloc((size_t)H_ * H_ * 2);
  _Float16* wt_o   = (_Float16*)alloc((size_t)H_ * H_ * 2);
  _Float16* wt_w1  = (_Float16*)alloc((size_t)FF_ * H_ * 2);
  _Float16* wt_w2  = (_Float16*)alloc((size_t)H_ * FF_ * 2);
  // activations
  _Float16* cat2   = (_Float16*)alloc((size_t)NTOK * 2 * H_ * 2);
  _Float16* cat4   = (_Float16*)alloc((size_t)NTOK * 4 * H_ * 2);
  float*    seq    = (float*)   alloc((size_t)NTOK * H_ * 4);
  float*    cell   = (float*)   alloc((size_t)B_ * L_ * H_ * 4);
  _Float16* cellh  = (_Float16*)alloc((size_t)B_ * L_ * H_ * 2);
  float*    cellout= (float*)   alloc((size_t)B_ * L_ * H_ * 4);
  float*    xbuf   = (float*)   alloc((size_t)NTOK * H_ * 4);   // reused gemm f32 out
  float*    xf32   = (float*)   alloc((size_t)NTOK * H_ * 4);
  _Float16* xh     = (_Float16*)alloc((size_t)NTOK * H_ * 2);
  _Float16* qb     = (_Float16*)alloc((size_t)NTOK * H_ * 2);
  _Float16* kb     = (_Float16*)alloc((size_t)NTOK * H_ * 2);
  _Float16* vb     = (_Float16*)alloc((size_t)NTOK * H_ * 2);
  _Float16* vt     = (_Float16*)alloc((size_t)NTOK * H_ * 2);
  _Float16* ctx    = (_Float16*)alloc((size_t)NTOK * H_ * 2);
  float*    hf32   = (float*)   alloc((size_t)NTOK * H_ * 4);
  _Float16* hh     = (_Float16*)alloc((size_t)NTOK * H_ * 2);
  _Float16* ffh    = (_Float16*)alloc((size_t)NTOK * FF_ * 2);
  (void)ws_size; (void)n_in; (void)in_sizes; (void)out_size;

  // ---- weight prep ----
  transpose_cast_kernel<<<ew_grid((size_t)2*H_*H_), 256, 0, stream>>>(r_dpw, wt_rdp, 2*H_, H_);
  transpose_cast_kernel<<<ew_grid((size_t)H_*H_),   256, 0, stream>>>(r_lw,  wt_rl,  H_,   H_);
  transpose_cast_kernel<<<ew_grid((size_t)2*H_*H_), 256, 0, stream>>>(c_dpw, wt_cdp, 2*H_, H_);
  transpose_cast_kernel<<<ew_grid((size_t)H_*H_),   256, 0, stream>>>(c_lw,  wt_cl,  H_,   H_);
  transpose_cast_kernel<<<ew_grid((size_t)4*H_*H_), 256, 0, stream>>>(dp_w,  wt_dp,  4*H_, H_);
  transpose_cast_kernel<<<ew_grid((size_t)H_*H_),   256, 0, stream>>>(wq,    wt_q,   H_,   H_);
  transpose_cast_kernel<<<ew_grid((size_t)H_*H_),   256, 0, stream>>>(wk,    wt_k,   H_,   H_);
  transpose_cast_kernel<<<ew_grid((size_t)H_*H_),   256, 0, stream>>>(wv,    wt_v,   H_,   H_);
  transpose_cast_kernel<<<ew_grid((size_t)H_*H_),   256, 0, stream>>>(wo,    wt_o,   H_,   H_);
  transpose_cast_kernel<<<ew_grid((size_t)H_*FF_),  256, 0, stream>>>(w1,    wt_w1,  H_,   FF_);
  transpose_cast_kernel<<<ew_grid((size_t)FF_*H_),  256, 0, stream>>>(w2,    wt_w2,  FF_,  H_);

  // ---- concat(former, hidden) into cat2 (and cat4 cols [0,2H)) ----
  concat2_kernel<<<ew_grid((size_t)NTOK*H_), 256, 0, stream>>>(former, hidden, cat2, cat4);

  const size_t NH_elems = (size_t)NTOK * H_;
  const size_t cell_elems = (size_t)B_ * L_ * H_;

  // ---- row table adapter ----
  gemm_wmma_kernel<<<dim3(NTOK/64, H_/64), 128, 0, stream>>>(cat2, wt_rdp, r_dpb, seq,
                                                             NTOK, H_, 2*H_, /*gelu*/1);
  zero_f32_kernel<<<ew_grid(cell_elems), 256, 0, stream>>>(cell, cell_elems);
  scatter_add_kernel<<<ew_grid(NH_elems), 256, 0, stream>>>(seq, row_ids, cell);
  cast_f16_kernel<<<ew_grid(cell_elems), 256, 0, stream>>>(cell, cellh, cell_elems);
  gemm_wmma_kernel<<<dim3((B_*L_)/64, H_/64), 128, 0, stream>>>(cellh, wt_rl, r_lb, cellout,
                                                                B_*L_, H_, H_, /*gelu*/1);
  gather_kernel<<<ew_grid(NH_elems), 256, 0, stream>>>(cellout, row_ids, col_ids, cat4, 2*H_);

  // ---- col table adapter ----
  gemm_wmma_kernel<<<dim3(NTOK/64, H_/64), 128, 0, stream>>>(cat2, wt_cdp, c_dpb, seq,
                                                             NTOK, H_, 2*H_, /*gelu*/1);
  zero_f32_kernel<<<ew_grid(cell_elems), 256, 0, stream>>>(cell, cell_elems);
  scatter_add_kernel<<<ew_grid(NH_elems), 256, 0, stream>>>(seq, col_ids, cell);
  cast_f16_kernel<<<ew_grid(cell_elems), 256, 0, stream>>>(cell, cellh, cell_elems);
  gemm_wmma_kernel<<<dim3((B_*L_)/64, H_/64), 128, 0, stream>>>(cellh, wt_cl, c_lb, cellout,
                                                                B_*L_, H_, H_, /*gelu*/1);
  gather_kernel<<<ew_grid(NH_elems), 256, 0, stream>>>(cellout, col_ids, col_ids, cat4, 3*H_);

  // ---- dp projection + LN ----
  gemm_wmma_kernel<<<dim3(NTOK/64, H_/64), 128, 0, stream>>>(cat4, wt_dp, dp_b, xbuf,
                                                             NTOK, H_, 4*H_, /*gelu*/1);
  add_ln_kernel<<<NTOK, 256, 0, stream>>>(former, xbuf, ln_g, ln_b, xf32, xh);

  // ---- QKV ----
  gemm_wmma_kernel<<<dim3(NTOK/64, H_/64), 128, 0, stream>>>(xh, wt_q, bq, qb,
                                                             NTOK, H_, H_, /*f16 out*/2);
  gemm_wmma_kernel<<<dim3(NTOK/64, H_/64), 128, 0, stream>>>(xh, wt_k, bk, kb,
                                                             NTOK, H_, H_, 2);
  gemm_wmma_kernel<<<dim3(NTOK/64, H_/64), 128, 0, stream>>>(xh, wt_v, bv, vb,
                                                             NTOK, H_, H_, 2);
  vtranspose_kernel<<<ew_grid(NH_elems), 256, 0, stream>>>(vb, vt);

  // ---- attention ----
  attention_kernel<<<dim3(B_ * NH_ * (S_/16)), 32, 0, stream>>>(qb, kb, vt, smask, ctx);

  // ---- output projection + LN ----
  gemm_wmma_kernel<<<dim3(NTOK/64, H_/64), 128, 0, stream>>>(ctx, wt_o, bo, xbuf,
                                                             NTOK, H_, H_, 0);
  add_ln_kernel<<<NTOK, 256, 0, stream>>>(xf32, xbuf, ln1_g, ln1_b, hf32, hh);

  // ---- FFN + final LN ----
  gemm_wmma_kernel<<<dim3(NTOK/64, FF_/64), 128, 0, stream>>>(hh, wt_w1, b1, ffh,
                                                              NTOK, FF_, H_, /*gelu+f16*/3);
  gemm_wmma_kernel<<<dim3(NTOK/64, H_/64), 128, 0, stream>>>(ffh, wt_w2, b2, xbuf,
                                                             NTOK, H_, FF_, 0);
  add_ln_kernel<<<NTOK, 256, 0, stream>>>(hf32, xbuf, ln2_g, ln2_b, (float*)d_out, nullptr);
}
